// MoELayer_27298812133635
// MI455X (gfx1250) — matrix operand order repaired and checked by reading, not compiled, not run on hardware
//
#include <hip/hip_runtime.h>
#include <cstdint>
#include <cstddef>

// MoE layer: D=1024, F=1024, E=8 routed experts (top-2) + 1 shared expert.
// N = 2*2048 = 4096 tokens. Shared expert is treated as expert index 8 with
// an identity token list and combine weight 1.0.
#define DD 1024
#define FF 1024
#define EE 8
#define NE 9           // EE routed + 1 shared
#define NN 4096
#define MT 64          // tokens per workgroup tile (4 m-tiles of 16)

typedef __attribute__((ext_vector_type(16))) __bf16 v16bf;
typedef __attribute__((ext_vector_type(8)))  float  v8f;

union Frag {
    v16bf v;
    uint4 q[2];
};

// round-to-nearest-even fp32 -> bf16 (bit pattern in ushort)
__device__ inline unsigned short f2bf(float f) {
    unsigned u = __float_as_uint(f);
    u += 0x7FFFu + ((u >> 16) & 1u);
    return (unsigned short)(u >> 16);
}

// A-fragment (16x32 bf16) for v_wmma_f32_16x16x32_bf16.
// Lane L holds row M = L&15. lanes 0-15: K = k0..k0+15;
// lanes 16-31: K = k0+16..k0+31. Two b128 loads.
__device__ inline v16bf load_frag_a(const unsigned short* __restrict__ row, int k0, int hi) {
    Frag f;
    const uint4* p = reinterpret_cast<const uint4*>(row + k0 + (hi ? 16 : 0));
    f.q[0] = p[0];       // K = k0 + 16*hi .. +7
    f.q[1] = p[1];       // K = k0 + 16*hi + 8 .. +15
    return f.v;
}

// B-fragment (32x16 bf16): lane holds column n = lane&15, 16 consecutive K
// starting at k0 + 16*hi -> 32 contiguous bytes in a row-major weight row.
__device__ inline v16bf load_frag_b(const unsigned short* __restrict__ p16) {
    Frag f;
    const uint4* p = reinterpret_cast<const uint4*>(p16);
    f.q[0] = p[0];
    f.q[1] = p[1];
    return f.v;
}

__device__ inline v8f wmma_bf16(v16bf a, v16bf b, v8f c) {
    return __builtin_amdgcn_wmma_f32_16x16x32_bf16(false, a, false, b, (short)0, c,
                                                   false, false);
}

// ---------------------------------------------------------------- prep ----
__global__ void cvt_kernel(const float* __restrict__ src,
                           unsigned short* __restrict__ dst, int n) {
    int i = blockIdx.x * 256 + threadIdx.x;
    int stride = gridDim.x * 256;
    for (; i < n; i += stride) dst[i] = f2bf(src[i]);
}

// zero routed-expert counts; build identity list for the shared expert
__global__ void init_kernel(int* __restrict__ cnt, int* __restrict__ list,
                            float* __restrict__ wlist) {
    int i = blockIdx.x * 256 + threadIdx.x;
    if (i < EE) cnt[i] = 0;
    if (i == 0) cnt[EE] = NN;
    if (i < NN) { list[EE * NN + i] = i; wlist[EE * NN + i] = 1.0f; }
}

// -------------------------------------------------------------- router ----
// one wave (32 lanes) per token: 8 router dots, softmax, top-2 + renorm,
// append (token, weight) to compacted per-expert lists; convert x to bf16.
__global__ __launch_bounds__(256) void router_kernel(
    const float* __restrict__ x, const float* __restrict__ Wr,
    unsigned short* __restrict__ xbf,
    int* __restrict__ cnt, int* __restrict__ list, float* __restrict__ wlist) {
    const int tid  = threadIdx.x;
    const int wave = tid >> 5, lane = tid & 31;
    const int n0   = blockIdx.x * 8;
    const int n    = n0 + wave;

    float s[EE];
#pragma unroll
    for (int e = 0; e < EE; e++) s[e] = 0.f;
    const float* xr = x + (size_t)n * DD;
    for (int j = lane; j < DD; j += 32) {
        float xv = xr[j];
#pragma unroll
        for (int e = 0; e < EE; e++) s[e] += xv * Wr[e * DD + j];
    }
#pragma unroll
    for (int off = 16; off > 0; off >>= 1) {
#pragma unroll
        for (int e = 0; e < EE; e++) s[e] += __shfl_xor(s[e], off, 32);
    }
    if (lane == 0) {
        float mx = s[0];
#pragma unroll
        for (int e = 1; e < EE; e++) mx = fmaxf(mx, s[e]);
        float p[EE], sum = 0.f;
#pragma unroll
        for (int e = 0; e < EE; e++) { p[e] = __expf(s[e] - mx); sum += p[e]; }
        float inv = 1.f / sum;
        int i1 = 0; float v1 = -1.f;
#pragma unroll
        for (int e = 0; e < EE; e++) {
            float sc = p[e] * inv; p[e] = sc;
            if (sc > v1) { v1 = sc; i1 = e; }
        }
        int i2 = 0; float v2 = -2.f;
#pragma unroll
        for (int e = 0; e < EE; e++) {
            if (e != i1 && p[e] > v2) { v2 = p[e]; i2 = e; }
        }
        float norm = 1.f / (v1 + v2 + 1e-8f);
        int pos1 = atomicAdd(&cnt[i1], 1);
        list[i1 * NN + pos1] = n; wlist[i1 * NN + pos1] = v1 * norm;
        int pos2 = atomicAdd(&cnt[i2], 1);
        list[i2 * NN + pos2] = n; wlist[i2 * NN + pos2] = v2 * norm;
    }
    const float* xb = x + (size_t)n0 * DD;
    unsigned short* ob = xbf + (size_t)n0 * DD;
    for (int i = tid; i < 8 * DD; i += 256) ob[i] = f2bf(xb[i]);
}

// ----------------------------------------------------- gate/up -> h ----
// grid (tile64, expert 0..8). 64 gathered tokens per workgroup; each wave owns
// a 128-wide F slice; every B-fragment is shared by 4 m-tiles (8 WMMA / 2 B
// loads), cutting L2 weight traffic 4x vs a 16-token tile.
__global__ __launch_bounds__(256) void moe_gateup_kernel(
    const unsigned short* __restrict__ xbf,
    const unsigned short* __restrict__ Wg,   // [E][F][D] bf16
    const unsigned short* __restrict__ Wu,
    const unsigned short* __restrict__ Sg,   // [F][D] bf16
    const unsigned short* __restrict__ Su,
    const int* __restrict__ cnt, const int* __restrict__ list,
    const float* __restrict__ wlist,
    unsigned short* __restrict__ hbuf) {     // [NE][NN][F] bf16 (by slot)
    __shared__ int   stok[MT];
    __shared__ float swgt[MT];

    const int e = blockIdx.y;
    const int c = cnt[e];
    const int tile = blockIdx.x;
    if (tile * MT >= c) return;                       // uniform early-exit

    const int tid = threadIdx.x;
    if (tid < MT) {
        int idx = tile * MT + tid;
        if (idx < c) { stok[tid] = list[e * NN + idx]; swgt[tid] = wlist[e * NN + idx]; }
        else         { stok[tid] = 0;                  swgt[tid] = 0.f; }   // padded
    }
    __syncthreads();

    const int wave = tid >> 5, lane = tid & 31;
    const int hi = lane >> 4, nl = lane & 15;
    const int wbase = wave * 128;

    const unsigned short* WgE = (e < EE) ? (Wg + (size_t)e * FF * DD) : Sg;
    const unsigned short* WuE = (e < EE) ? (Wu + (size_t)e * FF * DD) : Su;

    const unsigned short* arow[4];
#pragma unroll
    for (int mt = 0; mt < 4; mt++)
        arow[mt] = xbf + (size_t)stok[mt * 16 + nl] * DD;

    for (int nt = 0; nt < 8; ++nt) {
        const int n0 = wbase + nt * 16;
        const v8f z = {0.f,0.f,0.f,0.f,0.f,0.f,0.f,0.f};
        v8f accg[4] = {z, z, z, z};
        v8f accu[4] = {z, z, z, z};
        const unsigned short* bg = WgE + (size_t)(n0 + nl) * DD + hi * 16;
        const unsigned short* bu = WuE + (size_t)(n0 + nl) * DD + hi * 16;
        for (int k0 = 0; k0 < DD; k0 += 32) {
            __builtin_prefetch(bg + k0 + 512, 0, 1);
            __builtin_prefetch(bu + k0 + 512, 0, 1);
            v16bf b0 = load_frag_b(bg + k0);
            v16bf b1 = load_frag_b(bu + k0);
#pragma unroll
            for (int mt = 0; mt < 4; mt++) {
                v16bf a = load_frag_a(arow[mt], k0, hi);
                accg[mt] = wmma_bf16(a, b0, accg[mt]);
                accu[mt] = wmma_bf16(a, b1, accu[mt]);
            }
        }
#pragma unroll
        for (int mt = 0; mt < 4; mt++) {
#pragma unroll
            for (int r = 0; r < 8; r++) {
                int m = r + hi * 8;
                float g = accg[mt][r], u = accu[mt][r];
                float h = (g / (1.f + __expf(-g))) * u * swgt[mt * 16 + m];  // SiLU*up*w
                size_t slot = (size_t)tile * MT + mt * 16 + m;
                hbuf[((size_t)e * NN + slot) * FF + n0 + nl] = f2bf(h);
            }
        }
    }
}

// --------------------------------------------------- down-proj scatter ----
__global__ __launch_bounds__(256) void moe_down_kernel(
    const unsigned short* __restrict__ hbuf,   // [NE][NN][F] bf16
    const unsigned short* __restrict__ Wd,     // [E][D][F] bf16
    const unsigned short* __restrict__ Sd,     // [D][F] bf16
    const int* __restrict__ cnt, const int* __restrict__ list,
    float* __restrict__ out) {
    __shared__ int stok[MT];

    const int e = blockIdx.y;
    const int c = cnt[e];
    const int tile = blockIdx.x;
    if (tile * MT >= c) return;

    const int tid = threadIdx.x;
    if (tid < MT) {
        int idx = tile * MT + tid;
        stok[tid] = (idx < c) ? list[e * NN + idx] : 0;   // padded h rows are zero
    }
    __syncthreads();

    const int wave = tid >> 5, lane = tid & 31;
    const int hi = lane >> 4, nl = lane & 15;
    const int wbase = wave * 128;

    const unsigned short* WdE = (e < EE) ? (Wd + (size_t)e * DD * FF) : Sd;

    const unsigned short* hrow[4];
#pragma unroll
    for (int mt = 0; mt < 4; mt++)
        hrow[mt] = hbuf + ((size_t)e * NN + (size_t)tile * MT + mt * 16 + nl) * FF;

    for (int nt = 0; nt < 8; ++nt) {
        const int d0 = wbase + nt * 16;
        const v8f z = {0.f,0.f,0.f,0.f,0.f,0.f,0.f,0.f};
        v8f acc[4] = {z, z, z, z};
        const unsigned short* bd = WdE + (size_t)(d0 + nl) * FF + hi * 16;
        for (int k0 = 0; k0 < FF; k0 += 32) {
            __builtin_prefetch(bd + k0 + 512, 0, 1);
            v16bf b = load_frag_b(bd + k0);
#pragma unroll
            for (int mt = 0; mt < 4; mt++) {
                v16bf a = load_frag_a(hrow[mt], k0, hi);
                acc[mt] = wmma_bf16(a, b, acc[mt]);
            }
        }
#pragma unroll
        for (int mt = 0; mt < 4; mt++) {
#pragma unroll
            for (int r = 0; r < 8; r++) {
                int m = r + hi * 8;
                atomicAdd(&out[(size_t)stok[mt * 16 + m] * DD + d0 + nl], acc[mt][r]);
            }
        }
    }
}

// ---------------------------------------------------------------- host ----
extern "C" void kernel_launch(void* const* d_in, const int* in_sizes, int n_in,
                              void* d_out, int out_size, void* d_ws, size_t ws_size,
                              hipStream_t stream) {
    (void)in_sizes; (void)n_in; (void)ws_size;
    const float* x  = (const float*)d_in[0];
    const float* Wr = (const float*)d_in[1];
    const float* Wg = (const float*)d_in[2];
    const float* Wu = (const float*)d_in[3];
    const float* Wd = (const float*)d_in[4];
    const float* Sg = (const float*)d_in[5];
    const float* Su = (const float*)d_in[6];
    const float* Sd = (const float*)d_in[7];
    float* out = (float*)d_out;

    char* ws = (char*)d_ws;
    size_t off = 0;
    unsigned short* xbf = (unsigned short*)(ws + off); off += (size_t)NN * DD * 2;
    unsigned short* Wgb = (unsigned short*)(ws + off); off += (size_t)EE * FF * DD * 2;
    unsigned short* Wub = (unsigned short*)(ws + off); off += (size_t)EE * FF * DD * 2;
    unsigned short* Wdb = (unsigned short*)(ws + off); off += (size_t)EE * DD * FF * 2;
    unsigned short* Sgb = (unsigned short*)(ws + off); off += (size_t)FF * DD * 2;
    unsigned short* Sub = (unsigned short*)(ws + off); off += (size_t)FF * DD * 2;
    unsigned short* Sdb = (unsigned short*)(ws + off); off += (size_t)DD * FF * 2;
    int*   cnt   = (int*)(ws + off);            off += 64;
    int*   list  = (int*)(ws + off);            off += (size_t)NE * NN * 4;
    float* wlist = (float*)(ws + off);          off += (size_t)NE * NN * 4;
    unsigned short* hbuf = (unsigned short*)(ws + off);
    off += (size_t)NE * NN * FF * 2;

    // 1) weights fp32 -> bf16 (one HBM pass; bf16 copies live in the 192MB L2)
    cvt_kernel<<<2048, 256, 0, stream>>>(Wg, Wgb, EE * FF * DD);
    cvt_kernel<<<2048, 256, 0, stream>>>(Wu, Wub, EE * FF * DD);
    cvt_kernel<<<2048, 256, 0, stream>>>(Wd, Wdb, EE * DD * FF);
    cvt_kernel<<<512,  256, 0, stream>>>(Sg, Sgb, FF * DD);
    cvt_kernel<<<512,  256, 0, stream>>>(Su, Sub, FF * DD);
    cvt_kernel<<<512,  256, 0, stream>>>(Sd, Sdb, DD * FF);

    // 2) routing + x -> bf16; identity list for shared expert
    init_kernel<<<NN / 256, 256, 0, stream>>>(cnt, list, wlist);
    router_kernel<<<NN / 8, 256, 0, stream>>>(x, Wr, xbf, cnt, list, wlist);

    // 3) zero out, then gate/up -> h, then down-proj atomic scatter
    hipMemsetAsync(d_out, 0, (size_t)out_size * sizeof(float), stream);
    moe_gateup_kernel<<<dim3(NN / MT, NE), 256, 0, stream>>>(
        xbf, Wgb, Wub, Sgb, Sub, cnt, list, wlist, hbuf);
    moe_down_kernel<<<dim3(NN / MT, NE), 256, 0, stream>>>(
        hbuf, Wdb, Sdb, cnt, list, out);
}
